// GraphSAGEModel_18107582119954
// MI455X (gfx1250) — matrix operand order repaired and checked by reading, not compiled, not run on hardware
//
#include <hip/hip_runtime.h>
#include <hip/hip_bf16.h>

// ---------------------------------------------------------------------------
// GraphSAGE (4x SAGEConv + mean-pool + 4x Linear) for MI455X / gfx1250.
//   - scatter-mean via global f32 atomics (L2-resident irregular part)
//   - node GEMMs via v_wmma_f32_16x16x32_bf16, 16 nodes per wave32
//   - weights pre-swizzled to WMMA fragment order (bf16) once, staged to LDS
//     as raw 128-bit copies, fragments loaded as contiguous ds_load_b128
// ---------------------------------------------------------------------------

#define NN 100000     // N_NODES
#define NE 1600000    // N_EDGES
#define NG 1024       // N_GRAPHS
#define INC 64        // IN_CH
#define HID 32
#define H4 128
#define H2 64

typedef __attribute__((ext_vector_type(16))) __bf16 v16bf;
typedef __attribute__((ext_vector_type(8)))  float  v8f;

union BF16x16 { v16bf v; __bf16 e[16]; };
union F32x8   { v8f   v; float  e[8];  };

__device__ __forceinline__ void atomic_add_f32(float* p, float v) {
  __hip_atomic_fetch_add(p, v, __ATOMIC_RELAXED, __HIP_MEMORY_SCOPE_AGENT);
}

// ---------------------------------------------------------------------------
__global__ void zero_f32(float* __restrict__ p, int n) {
  int i = blockIdx.x * blockDim.x + threadIdx.x;
  if (i < n) p[i] = 0.0f;
}

__global__ void degree_kernel(const int* __restrict__ dst, float* __restrict__ deg) {
  int e = blockIdx.x * blockDim.x + threadIdx.x;
  if (e >= NE) return;
  unsigned d = (unsigned)dst[e];
  if (d < NN) atomic_add_f32(&deg[d], 1.0f);
}

__global__ void invdeg_kernel(float* __restrict__ deg) {
  int i = blockIdx.x * blockDim.x + threadIdx.x;
  if (i < NN) deg[i] = 1.0f / fmaxf(deg[i], 1.0f);
}

// ---------------------------------------------------------------------------
// Pre-swizzle a row-major [F_IN, F_OUT] f32 weight matrix into bf16 WMMA
// B-fragment order:
//   packed[((ks*NTILES + nt)*32 + lane)*16 + j]
//     = W[32*ks + (lane>>4)*16 + j][nt*16 + (lane&15)]
// so each lane's 16 B-values are one contiguous 32-byte run.
// ---------------------------------------------------------------------------
template <int F_IN, int F_OUT>
__global__ void pack_weights(const float* __restrict__ w, __bf16* __restrict__ packed) {
  constexpr int NTILES = F_OUT / 16;
  int i = blockIdx.x * blockDim.x + threadIdx.x;
  if (i >= F_IN * F_OUT) return;
  const int j    = i & 15;
  const int lane = (i >> 4) & 31;
  const int tile = i >> 9;
  const int ks   = tile / NTILES;
  const int nt   = tile - ks * NTILES;
  const int k    = 32 * ks + (lane >> 4) * 16 + j;
  const int n    = nt * 16 + (lane & 15);
  packed[i] = (__bf16)w[k * F_OUT + n];
}

// ---------------------------------------------------------------------------
// one thread per (edge, 8-channel group): 1 index pair read, 2x b128 gather,
// 8 global f32 atomic adds
// ---------------------------------------------------------------------------
template <int F>
__global__ void scatter_sum_kernel(const float* __restrict__ x,
                                   const int* __restrict__ src,
                                   const int* __restrict__ dst,
                                   float* __restrict__ agg) {
  constexpr int C8 = F / 8;
  long long t = (long long)blockIdx.x * blockDim.x + threadIdx.x;
  if (t >= (long long)NE * C8) return;
  int e = (int)(t / C8);
  int c = (int)(t % C8) * 8;
  unsigned s = (unsigned)src[e], d = (unsigned)dst[e];
  if (s >= NN || d >= NN) return;
  const float4 v0 = *(const float4*)(x + (size_t)s * F + c);
  const float4 v1 = *(const float4*)(x + (size_t)s * F + c + 4);
  float* o = agg + (size_t)d * F + c;
  atomic_add_f32(o + 0, v0.x);
  atomic_add_f32(o + 1, v0.y);
  atomic_add_f32(o + 2, v0.z);
  atomic_add_f32(o + 3, v0.w);
  atomic_add_f32(o + 4, v1.x);
  atomic_add_f32(o + 5, v1.y);
  atomic_add_f32(o + 6, v1.z);
  atomic_add_f32(o + 7, v1.w);
}

// ---------------------------------------------------------------------------
// out[16n : 16n+16, :] = relu( (agg*invdeg) @ WL + bl + X @ WR )
// One wave32 per 16-node tile. bf16 WMMA, f32 accumulate.
// A layout (16-bit 16x32): lane L: row M = L&15; khi = L>>4 selects K halves
// B fragments come pre-swizzled; per-lane load is one contiguous 32B run.
// C/D layout (f32 16x16): lane L: col N = L&15; VGPR r -> row M = (L>>4)*8+r
// ---------------------------------------------------------------------------
template <int F_IN, int F_OUT>
__global__ __launch_bounds__(256) void sage_gemm(
    const float* __restrict__ agg, const float* __restrict__ invdeg,
    const float* __restrict__ xin,
    const __bf16* __restrict__ pwl, const float* __restrict__ bl,
    const __bf16* __restrict__ pwr,
    float* __restrict__ out) {
  constexpr int KSTEPS = F_IN / 32;
  constexpr int NTILES = F_OUT / 16;
  constexpr int NELEM  = F_IN * F_OUT;

  __shared__ __bf16 sWL[NELEM];
  __shared__ __bf16 sWR[NELEM];
  {
    const uint4* gl = (const uint4*)pwl;
    const uint4* gr = (const uint4*)pwr;
    uint4* sl = (uint4*)sWL;
    uint4* sr = (uint4*)sWR;
    for (int i = threadIdx.x; i < NELEM / 8; i += blockDim.x) {
      sl[i] = gl[i];
      sr[i] = gr[i];
    }
  }
  __syncthreads();

  const int lane  = threadIdx.x & 31;
  const int wave  = threadIdx.x >> 5;
  const int mtile = blockIdx.x * 8 + wave;
  if (mtile * 16 >= NN) return;

  const int   mrow  = lane & 15;
  const int   khi   = lane >> 4;
  const int   row   = mtile * 16 + mrow;
  const float scale = invdeg[row];

  BF16x16 Aagg[KSTEPS], Ax[KSTEPS];
#pragma unroll
  for (int ks = 0; ks < KSTEPS; ++ks) {
    const int kg0 = 32 * ks + khi * 16;
    const int kg1 = 32 * ks + khi * 16 + 8;
    const float4 a0 = *(const float4*)(agg + (size_t)row * F_IN + kg0);
    const float4 a1 = *(const float4*)(agg + (size_t)row * F_IN + kg0 + 4);
    const float4 a2 = *(const float4*)(agg + (size_t)row * F_IN + kg1);
    const float4 a3 = *(const float4*)(agg + (size_t)row * F_IN + kg1 + 4);
    const float4 x0 = *(const float4*)(xin + (size_t)row * F_IN + kg0);
    const float4 x1 = *(const float4*)(xin + (size_t)row * F_IN + kg0 + 4);
    const float4 x2 = *(const float4*)(xin + (size_t)row * F_IN + kg1);
    const float4 x3 = *(const float4*)(xin + (size_t)row * F_IN + kg1 + 4);
    const float ta[16] = {a0.x, a0.y, a0.z, a0.w, a1.x, a1.y, a1.z, a1.w,
                          a2.x, a2.y, a2.z, a2.w, a3.x, a3.y, a3.z, a3.w};
    const float tx[16] = {x0.x, x0.y, x0.z, x0.w, x1.x, x1.y, x1.z, x1.w,
                          x2.x, x2.y, x2.z, x2.w, x3.x, x3.y, x3.z, x3.w};
#pragma unroll
    for (int j = 0; j < 16; ++j) {
      Aagg[ks].e[j] = (__bf16)(ta[j] * scale);
      Ax[ks].e[j]   = (__bf16)tx[j];
    }
  }

#pragma unroll
  for (int nt = 0; nt < NTILES; ++nt) {
    const int ncol = nt * 16 + mrow;
    F32x8 acc;
#pragma unroll
    for (int r = 0; r < 8; ++r) acc.e[r] = 0.0f;

#pragma unroll
    for (int ks = 0; ks < KSTEPS; ++ks) {
      const int tb = ((ks * NTILES + nt) * 32 + lane) * 16;
      BF16x16 Bl, Br;
      Bl.v = *(const v16bf*)(sWL + tb);
      Br.v = *(const v16bf*)(sWR + tb);
      acc.v = __builtin_amdgcn_wmma_f32_16x16x32_bf16(
          false, Aagg[ks].v, false, Bl.v, (short)0, acc.v, false, false);
      acc.v = __builtin_amdgcn_wmma_f32_16x16x32_bf16(
          false, Ax[ks].v, false, Br.v, (short)0, acc.v, false, false);
    }

    const float bias = bl[ncol];
#pragma unroll
    for (int r = 0; r < 8; ++r) {
      const int m = khi * 8 + r;
      const float v = acc.e[r] + bias;
      out[(size_t)(mtile * 16 + m) * F_OUT + ncol] = fmaxf(v, 0.0f);
    }
  }
}

// ---------------------------------------------------------------------------
__global__ void pool_kernel(const float* __restrict__ h,
                            const int* __restrict__ batch,
                            float* __restrict__ gsum, float* __restrict__ gcnt) {
  int t = blockIdx.x * blockDim.x + threadIdx.x;
  if (t >= NN * H2) return;
  int node = t >> 6;  // H2 == 64
  int c = t & 63;
  unsigned b = (unsigned)batch[node];
  if (b >= NG) return;
  atomic_add_f32(&gsum[b * H2 + c], h[(size_t)node * H2 + c]);
  if (c == 0) atomic_add_f32(&gcnt[b], 1.0f);
}

__global__ void mlp_kernel(const float* __restrict__ gsum, const float* __restrict__ gcnt,
                           const float* __restrict__ w1, const float* __restrict__ b1,
                           const float* __restrict__ w2, const float* __restrict__ b2,
                           const float* __restrict__ w3, const float* __restrict__ b3,
                           const float* __restrict__ w4, const float* __restrict__ b4,
                           float* __restrict__ out) {
  int g = blockIdx.x * blockDim.x + threadIdx.x;
  if (g >= NG) return;
  const float inv = 1.0f / fmaxf(gcnt[g], 1.0f);
  float v0[H2], v1[H2], v2[HID], v3[HID];
  for (int c = 0; c < H2; ++c) v0[c] = gsum[g * H2 + c] * inv;
  for (int o = 0; o < H2; ++o) {
    float s = b1[o];
    for (int k = 0; k < H2; ++k) s += v0[k] * w1[k * H2 + o];
    v1[o] = fmaxf(s, 0.0f);
  }
  for (int o = 0; o < HID; ++o) {
    float s = b2[o];
    for (int k = 0; k < H2; ++k) s += v1[k] * w2[k * HID + o];
    v2[o] = fmaxf(s, 0.0f);
  }
  for (int o = 0; o < HID; ++o) {
    float s = b3[o];
    for (int k = 0; k < HID; ++k) s += v2[k] * w3[k * HID + o];
    v3[o] = fmaxf(s, 0.0f);
  }
  float s = b4[0];
  for (int k = 0; k < HID; ++k) s += v3[k] * w4[k];
  out[g] = s;
}

// ---------------------------------------------------------------------------
extern "C" void kernel_launch(void* const* d_in, const int* in_sizes, int n_in,
                              void* d_out, int out_size, void* d_ws, size_t ws_size,
                              hipStream_t stream) {
  const float* x     = (const float*)d_in[0];
  const int*   ei    = (const int*)d_in[1];
  const int*   src   = ei;
  const int*   dst   = ei + NE;
  const int*   batch = (const int*)d_in[2];
  const float* c1_wl = (const float*)d_in[3];
  const float* c1_bl = (const float*)d_in[4];
  const float* c1_wr = (const float*)d_in[5];
  const float* c2_wl = (const float*)d_in[6];
  const float* c2_bl = (const float*)d_in[7];
  const float* c2_wr = (const float*)d_in[8];
  const float* c3_wl = (const float*)d_in[9];
  const float* c3_bl = (const float*)d_in[10];
  const float* c3_wr = (const float*)d_in[11];
  const float* c4_wl = (const float*)d_in[12];
  const float* c4_bl = (const float*)d_in[13];
  const float* c4_wr = (const float*)d_in[14];
  const float* l1w = (const float*)d_in[15];
  const float* l1b = (const float*)d_in[16];
  const float* l2w = (const float*)d_in[17];
  const float* l2b = (const float*)d_in[18];
  const float* l3w = (const float*)d_in[19];
  const float* l3b = (const float*)d_in[20];
  const float* l4w = (const float*)d_in[21];
  const float* l4b = (const float*)d_in[22];

  // workspace carve-out (256B aligned slabs)
  char* ws = (char*)d_ws;
  size_t off = 0;
  auto take_b = [&](size_t bytes) -> void* {
    void* p = ws + off;
    off += (bytes + 255) & ~(size_t)255;
    return p;
  };
  float* agg  = (float*)take_b((size_t)NN * H4 * 4);
  float* invd = (float*)take_b((size_t)NN * 4);
  float* hA   = (float*)take_b((size_t)NN * H4 * 4);
  float* hB   = (float*)take_b((size_t)NN * H4 * 4);
  float* gsum = (float*)take_b((size_t)NG * H2 * 4);
  float* gcnt = (float*)take_b((size_t)NG * 4);
  __bf16* pw1l = (__bf16*)take_b((size_t)INC * H4 * 2);
  __bf16* pw1r = (__bf16*)take_b((size_t)INC * H4 * 2);
  __bf16* pw2l = (__bf16*)take_b((size_t)H4 * H2 * 2);
  __bf16* pw2r = (__bf16*)take_b((size_t)H4 * H2 * 2);
  __bf16* pw3l = (__bf16*)take_b((size_t)H2 * H2 * 2);
  __bf16* pw3r = (__bf16*)take_b((size_t)H2 * H2 * 2);
  __bf16* pw4l = (__bf16*)take_b((size_t)H2 * H2 * 2);
  __bf16* pw4r = (__bf16*)take_b((size_t)H2 * H2 * 2);

  const dim3 blk(256);
  auto grid1 = [](long long n) { return dim3((unsigned)((n + 255) / 256)); };
  const int MT = (NN + 15) / 16;   // 6250 node tiles
  const dim3 gridG((MT + 7) / 8);  // 8 waves per block

  // pre-swizzle all weight matrices into WMMA fragment order (bf16)
  pack_weights<64, 128><<<grid1(64 * 128), blk, 0, stream>>>(c1_wl, pw1l);
  pack_weights<64, 128><<<grid1(64 * 128), blk, 0, stream>>>(c1_wr, pw1r);
  pack_weights<128, 64><<<grid1(128 * 64), blk, 0, stream>>>(c2_wl, pw2l);
  pack_weights<128, 64><<<grid1(128 * 64), blk, 0, stream>>>(c2_wr, pw2r);
  pack_weights<64, 64><<<grid1(64 * 64), blk, 0, stream>>>(c3_wl, pw3l);
  pack_weights<64, 64><<<grid1(64 * 64), blk, 0, stream>>>(c3_wr, pw3r);
  pack_weights<64, 64><<<grid1(64 * 64), blk, 0, stream>>>(c4_wl, pw4l);
  pack_weights<64, 64><<<grid1(64 * 64), blk, 0, stream>>>(c4_wr, pw4r);

  // degrees (shared by all layers)
  zero_f32<<<grid1(NN), blk, 0, stream>>>(invd, NN);
  degree_kernel<<<grid1(NE), blk, 0, stream>>>(dst, invd);
  invdeg_kernel<<<grid1(NN), blk, 0, stream>>>(invd);

  // layer 1: 64 -> 128
  zero_f32<<<grid1((long long)NN * INC), blk, 0, stream>>>(agg, NN * INC);
  scatter_sum_kernel<64><<<grid1((long long)NE * 8), blk, 0, stream>>>(x, src, dst, agg);
  sage_gemm<64, 128><<<gridG, blk, 0, stream>>>(agg, invd, x, pw1l, c1_bl, pw1r, hA);

  // layer 2: 128 -> 64
  zero_f32<<<grid1((long long)NN * H4), blk, 0, stream>>>(agg, NN * H4);
  scatter_sum_kernel<128><<<grid1((long long)NE * 16), blk, 0, stream>>>(hA, src, dst, agg);
  sage_gemm<128, 64><<<gridG, blk, 0, stream>>>(agg, invd, hA, pw2l, c2_bl, pw2r, hB);

  // layer 3: 64 -> 64
  zero_f32<<<grid1((long long)NN * H2), blk, 0, stream>>>(agg, NN * H2);
  scatter_sum_kernel<64><<<grid1((long long)NE * 8), blk, 0, stream>>>(hB, src, dst, agg);
  sage_gemm<64, 64><<<gridG, blk, 0, stream>>>(agg, invd, hB, pw3l, c3_bl, pw3r, hA);

  // layer 4: 64 -> 64
  zero_f32<<<grid1((long long)NN * H2), blk, 0, stream>>>(agg, NN * H2);
  scatter_sum_kernel<64><<<grid1((long long)NE * 8), blk, 0, stream>>>(hA, src, dst, agg);
  sage_gemm<64, 64><<<gridG, blk, 0, stream>>>(agg, invd, hA, pw4l, c4_bl, pw4r, hB);

  // global mean pool + MLP head
  zero_f32<<<grid1((long long)NG * H2), blk, 0, stream>>>(gsum, NG * H2);
  zero_f32<<<grid1(NG), blk, 0, stream>>>(gcnt, NG);
  pool_kernel<<<grid1((long long)NN * H2), blk, 0, stream>>>(hB, batch, gsum, gcnt);
  mlp_kernel<<<grid1(NG), blk, 0, stream>>>(gsum, gcnt, l1w, l1b, l2w, l2b,
                                            l3w, l3b, l4w, l4b, (float*)d_out);
}
